// RecipeFFN_26259430048184
// MI455X (gfx1250) — matrix operand order repaired and checked by reading, not compiled, not run on hardware
//
#include <hip/hip_runtime.h>
#include <hip/hip_bf16.h>
#include <math.h>

typedef __attribute__((ext_vector_type(16))) __bf16 v16bf;
typedef __attribute__((ext_vector_type(8)))  float  v8f;

#define NTOK 2048      // B*S
#define DIM  1024      // D
#define KSEL 8
#define NBAS 64
#define RANK 64
#define NR   4096      // NBAS*RANK
#define DFF  4096

#if __has_builtin(__builtin_amdgcn_global_load_async_to_lds_b128)
#define HAVE_ASYNC_LDS 1
#endif

typedef int v4i __attribute__((vector_size(4 * sizeof(int))));
typedef __attribute__((address_space(1))) v4i as1_v4i;
typedef __attribute__((address_space(3))) v4i as3_v4i;

static __device__ __forceinline__ __bf16 f2bf(float f) {
  unsigned u = __float_as_uint(f);
  unsigned r = (u + 0x7FFFu + ((u >> 16) & 1u)) >> 16;
  unsigned short s = (unsigned short)r;
  return __builtin_bit_cast(__bf16, s);
}

#ifdef HAVE_ASYNC_LDS
// 16B async copy global -> LDS (ASYNCcnt path, no VGPR round trip)
static __device__ __forceinline__ void async_cp16(void* lds, const void* glob) {
  as1_v4i* g = (as1_v4i*)(unsigned long long)glob;
  as3_v4i* l = (as3_v4i*)(unsigned)(unsigned long long)lds;  // low 32 bits = LDS offset
  __builtin_amdgcn_global_load_async_to_lds_b128(g, l, 0, 0);
}
static __device__ __forceinline__ void wait_async0() {
#if __has_builtin(__builtin_amdgcn_s_wait_asynccnt)
  __builtin_amdgcn_s_wait_asynccnt(0);
#else
  asm volatile("s_wait_asynccnt 0x0" ::: "memory");
#endif
}
#endif

// ---------------- token recipe: gather + softmax + weighted sum -> c[2048][64]
__global__ void k_token_recipe(const float* __restrict__ recipe,
                               const int* __restrict__ idx,
                               const float* __restrict__ wts,
                               float* __restrict__ c) {
  const int tok  = blockIdx.x;
  const int lane = threadIdx.x;            // 0..31 (wave32)
  float acc0 = 0.f, acc1 = 0.f;
  for (int k = 0; k < KSEL; ++k) {
    const int   ni = idx[tok * KSEL + k];
    const float wk = wts[tok * KSEL + k];
    const float* row = recipe + (size_t)ni * NBAS;
    float v0 = row[lane], v1 = row[lane + 32];
    float m = fmaxf(v0, v1);
    for (int off = 16; off >= 1; off >>= 1) m = fmaxf(m, __shfl_xor(m, off, 32));
    float e0 = expf(v0 - m), e1 = expf(v1 - m);
    float s = e0 + e1;
    for (int off = 16; off >= 1; off >>= 1) s += __shfl_xor(s, off, 32);
    float inv = wk / s;
    acc0 += e0 * inv;
    acc1 += e1 * inv;
  }
  c[tok * NBAS + lane]      = acc0;
  c[tok * NBAS + lane + 32] = acc1;
}

// ---------------- elementwise f32 -> bf16
__global__ void k_convert(const float* __restrict__ in, __bf16* __restrict__ out, int n) {
  int i = blockIdx.x * blockDim.x + threadIdx.x;
  if (i < n) out[i] = f2bf(in[i]);
}

// ---------------- basis_A[n][d][r] -> Bt1[(n*64+r)][d] (bf16)
__global__ void k_build_Bt1(const float* __restrict__ A, __bf16* __restrict__ Bt, int total) {
  int i = blockIdx.x * blockDim.x + threadIdx.x;
  if (i >= total) return;
  int d   = i & (DIM - 1);
  int col = i >> 10;                 // n*64 + r
  int n   = col >> 6;
  int r   = col & 63;
  Bt[(size_t)col * DIM + d] = f2bf(A[(size_t)n * (DIM * RANK) + (size_t)d * RANK + r]);
}

// ---------------- h[tok][r] = sum_n c[tok][n] * t[tok][n*64+r]
__global__ void k_reduce_h(const float* __restrict__ c, const float* __restrict__ t,
                           float* __restrict__ h) {
  const int tok = blockIdx.x;
  const int r   = threadIdx.x;       // 0..63
  const float* tp = t + (size_t)tok * NR;
  const float* cp = c + (size_t)tok * NBAS;
  float acc = 0.f;
  #pragma unroll 8
  for (int n = 0; n < NBAS; ++n) acc += cp[n] * tp[n * RANK + r];
  h[tok * RANK + r] = acc;
}

// ---------------- w[tok][(n,r)] = bf16(c[tok][n] * h[tok][r])
__global__ void k_build_w(const float* __restrict__ c, const float* __restrict__ h,
                          __bf16* __restrict__ w, int total) {
  int i = blockIdx.x * blockDim.x + threadIdx.x;
  if (i >= total) return;
  int tok = i >> 12;
  int nr  = i & 4095;
  w[i] = f2bf(c[tok * NBAS + (nr >> 6)] * h[tok * RANK + (nr & 63)]);
}

// ---------------- f32 src[R][C] -> bf16 dst[C][R] (LDS-tiled transpose)
__global__ void k_transpose(const float* __restrict__ src, __bf16* __restrict__ dst,
                            int R, int C) {
  __shared__ float tile[32][33];
  const int x  = blockIdx.x * 32 + threadIdx.x;   // col in src
  const int y0 = blockIdx.y * 32;                 // row base in src
  #pragma unroll
  for (int j = 0; j < 32; j += 8)
    tile[threadIdx.y + j][threadIdx.x] = src[(size_t)(y0 + threadIdx.y + j) * C + x];
  __syncthreads();
  const int xo  = y0 + threadIdx.x;               // col in dst
  const int yo0 = blockIdx.x * 32;                // row base in dst
  #pragma unroll
  for (int j = 0; j < 32; j += 8)
    dst[(size_t)(yo0 + threadIdx.y + j) * R + xo] = f2bf(tile[threadIdx.x][threadIdx.y + j]);
}

// ---------------- WMMA GEMM: C[M][N] = A[M][K] @ Bt[N][K]^T   (bf16 -> f32 acc)
// Double-buffered LDS fill via global_load_async_to_lds_b128 (ASYNCcnt overlap).
// EPI 0: store f32 ; EPI 1: exact-erf gelu -> bf16 ; EPI 2: +bias[col] -> f32
template<int EPI>
__global__ void __launch_bounds__(256)
k_gemm(const __bf16* __restrict__ A, const __bf16* __restrict__ Bt,
       float* __restrict__ Cf, __bf16* __restrict__ Cb,
       const float* __restrict__ bias, int M, int N, int K) {
  __shared__ __align__(16) __bf16 As[2][128 * 32];
  __shared__ __align__(16) __bf16 Bs[2][128 * 32];

  const int tid   = threadIdx.x;
  const int lane  = tid & 31;
  const int wid   = tid >> 5;
  const int waveM = wid >> 1;       // 0..3 -> 32 rows each
  const int waveN = wid & 1;        // 0..1 -> 64 cols each
  const int bm = blockIdx.y * 128;
  const int bn = blockIdx.x * 128;

  const int lrow  = tid >> 1;       // 0..127  (tile row loaded by this thread)
  const int lhalf = tid & 1;        // which 16-element K chunk
  const int loff  = lrow * 32 + lhalf * 16;

  const __bf16* gA = A  + (size_t)(bm + lrow) * K + lhalf * 16;
  const __bf16* gB = Bt + (size_t)(bn + lrow) * K + lhalf * 16;

  v8f acc[2][4];
  #pragma unroll
  for (int mt = 0; mt < 2; ++mt) {
    #pragma unroll
    for (int nt = 0; nt < 4; ++nt) {
      acc[mt][nt] = (v8f){0.f, 0.f, 0.f, 0.f, 0.f, 0.f, 0.f, 0.f};
    }
  }

  const int kTiles = K >> 5;

#ifdef HAVE_ASYNC_LDS
  // prologue: fill buffer 0 for tile 0
  async_cp16(&As[0][loff],     gA);
  async_cp16(&As[0][loff + 8], gA + 8);
  async_cp16(&Bs[0][loff],     gB);
  async_cp16(&Bs[0][loff + 8], gB + 8);
#endif

  for (int kt = 0; kt < kTiles; ++kt) {
    const int buf = kt & 1;

#ifdef HAVE_ASYNC_LDS
    wait_async0();                   // this wave's tile-kt DMA done
    __syncthreads();                 // everyone's tile-kt visible; buf^1 free to overwrite
    if (kt + 1 < kTiles) {           // kick next tile while we do the WMMAs below
      const __bf16* apn = gA + (size_t)(kt + 1) * 32;
      const __bf16* bpn = gB + (size_t)(kt + 1) * 32;
      async_cp16(&As[buf ^ 1][loff],     apn);
      async_cp16(&As[buf ^ 1][loff + 8], apn + 8);
      async_cp16(&Bs[buf ^ 1][loff],     bpn);
      async_cp16(&Bs[buf ^ 1][loff + 8], bpn + 8);
      __builtin_prefetch(apn + 32, 0, 1);   // tile kt+2 -> L2 (global_prefetch_b8)
      __builtin_prefetch(bpn + 32, 0, 1);
    }
#else
    const __bf16* ap = gA + (size_t)kt * 32;
    const __bf16* bp = gB + (size_t)kt * 32;
    uint4 a0 = ((const uint4*)ap)[0];
    uint4 a1 = ((const uint4*)ap)[1];
    uint4 b0 = ((const uint4*)bp)[0];
    uint4 b1 = ((const uint4*)bp)[1];
    __syncthreads();
    *(uint4*)(&As[buf][loff])     = a0;
    *(uint4*)(&As[buf][loff + 8]) = a1;
    *(uint4*)(&Bs[buf][loff])     = b0;
    *(uint4*)(&Bs[buf][loff + 8]) = b1;
    __syncthreads();
#endif

    // A fragment: lane half selects K {0..7,16..23} vs {8..15,24..31} (ISA 7.12.2)
    const int ah = lane >> 4;
    const int am = lane & 15;
    union { v16bf v; uint4 q[2]; } afr[2], bfr[4];
    #pragma unroll
    for (int mt = 0; mt < 2; ++mt) {
      const int m = waveM * 32 + mt * 16 + am;
      afr[mt].q[0] = *(const uint4*)(&As[buf][m * 32 + ah * 8]);
      afr[mt].q[1] = *(const uint4*)(&As[buf][m * 32 + 16 + ah * 8]);
    }
    // B fragment: column = lane&15, lanes 0-15 hold K {0..7,16..23}, lanes 16-31 K {8..15,24..31}
    const int bc = lane & 15;
    const int bk = (lane >> 4) * 8;
    #pragma unroll
    for (int nt = 0; nt < 4; ++nt) {
      const int n = waveN * 64 + nt * 16 + bc;
      bfr[nt].q[0] = *(const uint4*)(&Bs[buf][n * 32 + bk]);
      bfr[nt].q[1] = *(const uint4*)(&Bs[buf][n * 32 + 16 + bk]);
    }
    #pragma unroll
    for (int mt = 0; mt < 2; ++mt) {
      #pragma unroll
      for (int nt = 0; nt < 4; ++nt) {
        acc[mt][nt] = __builtin_amdgcn_wmma_f32_16x16x32_bf16(
            false, afr[mt].v, false, bfr[nt].v, (short)0, acc[mt][nt], false, false);
      }
    }
  }

  // D layout: lane col = lane&15, rows (lane>>4)*8 + j in VGPR j
  const int col = lane & 15;
  const int rh  = lane >> 4;
  #pragma unroll
  for (int mt = 0; mt < 2; ++mt) {
    #pragma unroll
    for (int nt = 0; nt < 4; ++nt) {
      const int gm0 = bm + waveM * 32 + mt * 16 + rh * 8;
      const int gn  = bn + waveN * 64 + nt * 16 + col;
      #pragma unroll
      for (int j = 0; j < 8; ++j) {
        float v = acc[mt][nt][j];
        if (EPI == 0) {
          Cf[(size_t)(gm0 + j) * N + gn] = v;
        } else if (EPI == 1) {
          float g = 0.5f * v * (1.0f + erff(v * 0.70710678118654752440f));
          Cb[(size_t)(gm0 + j) * N + gn] = f2bf(g);
        } else {
          Cf[(size_t)(gm0 + j) * N + gn] = v + bias[gn];
        }
      }
    }
  }
}

extern "C" void kernel_launch(void* const* d_in, const int* in_sizes, int n_in,
                              void* d_out, int out_size, void* d_ws, size_t ws_size,
                              hipStream_t stream) {
  const float* x   = (const float*)d_in[0];
  const int*   nid = (const int*)d_in[1];
  const float* nw  = (const float*)d_in[2];
  const float* rec = (const float*)d_in[3];
  const float* bA  = (const float*)d_in[4];
  const float* bB  = (const float*)d_in[5];
  const float* Wd  = (const float*)d_in[6];
  const float* Wb  = (const float*)d_in[7];
  float* out = (float*)d_out;

  char* ws = (char*)d_ws;
  const size_t MB = (size_t)1 << 20;
  float*  c   = (float*) (ws);                 // 0.5 MB
  float*  h   = (float*) (ws + MB / 2);        // 0.5 MB
  __bf16* Xb  = (__bf16*)(ws + 1 * MB);        // 4 MB
  __bf16* Bt1 = (__bf16*)(ws + 5 * MB);        // 8 MB (reused as Bt3)
  __bf16* w   = (__bf16*)(ws + 13 * MB);       // 16 MB
  __bf16* h2  = (__bf16*)(ws + 29 * MB);       // 16 MB
  float*  t   = (float*) (ws + 45 * MB);       // 32 MB (reused as Bt2)
  __bf16* Bt2 = (__bf16*)(ws + 45 * MB);
  __bf16* Bt3 = Bt1;

  // 1) token recipes
  k_token_recipe<<<NTOK, 32, 0, stream>>>(rec, nid, nw, c);
  // 2) x -> bf16
  k_convert<<<(NTOK * DIM) / 256, 256, 0, stream>>>(x, Xb, NTOK * DIM);
  // 3) basis_A -> Bt1[(n,r)][d]
  k_build_Bt1<<<(NR * DIM) / 256, 256, 0, stream>>>(bA, Bt1, NR * DIM);
  // 4) GEMM1: t[2048][4096] = Xb @ Bt1^T
  {
    dim3 g(NR / 128, NTOK / 128);
    k_gemm<0><<<g, 256, 0, stream>>>(Xb, Bt1, t, nullptr, nullptr, NTOK, NR, DIM);
  }
  // 5) h = sum_n c * t
  k_reduce_h<<<NTOK, RANK, 0, stream>>>(c, t, h);
  // 6) w = c (x) h  -> bf16
  k_build_w<<<(NTOK * NR) / 256, 256, 0, stream>>>(c, h, w, NTOK * NR);
  // 7) basis_B[nr][f] -> Bt2[f][nr] (overwrites t; t consumed in step 5)
  {
    dim3 tb(32, 8), tg(DFF / 32, NR / 32);
    k_transpose<<<tg, tb, 0, stream>>>(bB, Bt2, NR, DFF);
  }
  // 8) GEMM2 + gelu: h2[2048][4096] (bf16)
  {
    dim3 g(DFF / 128, NTOK / 128);
    k_gemm<1><<<g, 256, 0, stream>>>(w, Bt2, nullptr, h2, nullptr, NTOK, DFF, NR);
  }
  // 9) W_down -> bf16 (already Bt layout: [d][f])
  k_convert<<<(DIM * DFF) / 256, 256, 0, stream>>>(Wd, Bt3, DIM * DFF);
  // 10) GEMM3 + bias -> out f32
  {
    dim3 g(DIM / 128, NTOK / 128);
    k_gemm<2><<<g, 256, 0, stream>>>(h2, Bt3, out, nullptr, Wb, NTOK, DIM, DFF);
  }
}
